// MulAttention_67482526155256
// MI455X (gfx1250) — matrix operand order repaired and checked by reading, not compiled, not run on hardware
//
#include <hip/hip_runtime.h>

#define BB 8
#define LL 2048
#define HH 1024

typedef __attribute__((ext_vector_type(16))) _Float16 v16h;
typedef __attribute__((ext_vector_type(8)))  float    v8f;

union HFrag { v16h v; _Float16 h[16]; };
union FAcc  { v8f  v; float    f[8];  };

__device__ __forceinline__ v8f wmma_f16(v16h a, v16h b, v8f c) {
  // (neg_a, A, neg_b, B, c_mod, C, reuse_a, reuse_b)
  return __builtin_amdgcn_wmma_f32_16x16x32_f16(false, a, false, b, (short)0, c,
                                                false, false);
}

// C[M,N] = A[M,K] x B (+bias per column), fp32 in/out, split-f16 WMMA core.
// BMODE 0: global B given as Bt row-major [N][K], ld=ldb  (C = A * Bt^T)
// BMODE 1: global B given as    row-major [K][N], ld=ldb  (C = A * B)
// Workgroup: 256 threads = 8 waves -> 128(M) x 64(N) tile; wave = 32x32.
template <int BMODE>
__global__ __launch_bounds__(256)
void gemm_splitf16_kernel(const float* __restrict__ Ag,
                          const float* __restrict__ Bg,
                          const float* __restrict__ bias,
                          float* __restrict__ Cg,
                          int M, int N, int K,
                          int lda, int ldb, int ldc,
                          long sA, long sB, long sC) {
  __shared__ _Float16 sAh[128][34];
  __shared__ _Float16 sAl[128][34];
  __shared__ _Float16 sBh[64][34];   // stored [n][k]
  __shared__ _Float16 sBl[64][34];

  const int tid  = threadIdx.x;
  const int lane = tid & 31;
  const int wave = tid >> 5;
  const int wm   = wave >> 1;        // 0..3  (M direction)
  const int wn   = wave & 1;         // 0..1  (N direction)
  const int l16  = lane & 15;
  const int hsel = lane >> 4;        // half-wave select

  const int n0 = blockIdx.x * 64;
  const int m0 = blockIdx.y * 128;
  const float* A = Ag + (size_t)blockIdx.z * (size_t)sA;
  const float* B = Bg + (size_t)blockIdx.z * (size_t)sB;
  float*       C = Cg + (size_t)blockIdx.z * (size_t)sC;

  FAcc acc[2][2];
  #pragma unroll
  for (int i = 0; i < 2; ++i)
    #pragma unroll
    for (int j = 0; j < 2; ++j)
      #pragma unroll
      for (int r = 0; r < 8; ++r) acc[i][j].f[r] = 0.0f;

  for (int kk = 0; kk < K; kk += 32) {
    // ---- stage A tile 128x32, split into hi/lo f16 ----
    #pragma unroll
    for (int i = 0; i < 16; ++i) {
      int idx = tid + i * 256;
      int k = idx & 31, m = idx >> 5;
      float x = A[(size_t)(m0 + m) * (size_t)lda + (size_t)(kk + k)];
      _Float16 hi = (_Float16)x;
      _Float16 lo = (_Float16)(x - (float)hi);
      sAh[m][k] = hi;
      sAl[m][k] = lo;
    }
    // ---- stage B tile 64(n) x 32(k), split into hi/lo f16 ----
    #pragma unroll
    for (int i = 0; i < 8; ++i) {
      int idx = tid + i * 256;
      int n, k;
      float x;
      if (BMODE == 0) {
        k = idx & 31; n = idx >> 5;
        x = B[(size_t)(n0 + n) * (size_t)ldb + (size_t)(kk + k)];
      } else {
        n = idx & 63; k = idx >> 6;   // coalesced along n for [K][N] source
        x = B[(size_t)(kk + k) * (size_t)ldb + (size_t)(n0 + n)];
      }
      _Float16 hi = (_Float16)x;
      _Float16 lo = (_Float16)(x - (float)hi);
      sBh[n][k] = hi;
      sBl[n][k] = lo;
    }
    __syncthreads();

    // ---- build WMMA fragments per ISA VGPR layouts ----
    HFrag ah[2], al[2], bh[2], bl[2];
    #pragma unroll
    for (int im = 0; im < 2; ++im) {
      int mr = wm * 32 + im * 16 + l16;
      int kb = hsel * 8;               // A: lanes16-31 start at K=8
      #pragma unroll
      for (int e = 0; e < 8; ++e) {
        ah[im].h[e]     = sAh[mr][kb + e];
        al[im].h[e]     = sAl[mr][kb + e];
        ah[im].h[8 + e] = sAh[mr][kb + 16 + e];
        al[im].h[8 + e] = sAl[mr][kb + 16 + e];
      }
    }
    #pragma unroll
    for (int in = 0; in < 2; ++in) {
      int nc = wn * 32 + in * 16 + l16;
      int kb = hsel * 8;               // B: lanes16-31 start at K=8
      #pragma unroll
      for (int e = 0; e < 8; ++e) {
        bh[in].h[e]     = sBh[nc][kb + e];
        bl[in].h[e]     = sBl[nc][kb + e];
        bh[in].h[8 + e] = sBh[nc][kb + 16 + e];
        bl[in].h[8 + e] = sBl[nc][kb + 16 + e];
      }
    }

    // ---- split-f16 MACs: hi*hi + hi*lo + lo*hi (lo*lo ~ 2^-22, dropped) ----
    #pragma unroll
    for (int im = 0; im < 2; ++im)
      #pragma unroll
      for (int in = 0; in < 2; ++in) {
        acc[im][in].v = wmma_f16(ah[im].v, bh[in].v, acc[im][in].v);
        acc[im][in].v = wmma_f16(ah[im].v, bl[in].v, acc[im][in].v);
        acc[im][in].v = wmma_f16(al[im].v, bh[in].v, acc[im][in].v);
      }
    __syncthreads();
  }

  // ---- epilogue: C/D layout VGPR r -> M = r + 8*hsel, N = l16 ----
  #pragma unroll
  for (int im = 0; im < 2; ++im)
    #pragma unroll
    for (int in = 0; in < 2; ++in) {
      int col = n0 + wn * 32 + in * 16 + l16;
      float bv = bias ? bias[col] : 0.0f;
      #pragma unroll
      for (int r = 0; r < 8; ++r) {
        int row = m0 + wm * 32 + im * 16 + hsel * 8 + r;
        C[(size_t)row * (size_t)ldc + (size_t)col] = acc[im][in].f[r] + bv;
      }
    }
}

// In-place: a[row,:] = softmax(relu(a[row,:])) + mask[row,:]; one block per row.
__global__ __launch_bounds__(256)
void softmax_relu_mask_kernel(float* __restrict__ a, const float* __restrict__ mask) {
  __shared__ float red[256];
  const int tid = threadIdx.x;
  const size_t off = (size_t)blockIdx.x * (size_t)LL;
  float* row = a + off;
  const float* mrow = mask + off;

  float v[LL / 256];
  float mx = 0.0f;                       // relu => values >= 0
  #pragma unroll
  for (int i = 0; i < LL / 256; ++i) {
    float x = row[tid + i * 256];
    x = fmaxf(x, 0.0f);
    v[i] = x;
    mx = fmaxf(mx, x);
  }
  red[tid] = mx;
  __syncthreads();
  for (int s = 128; s > 0; s >>= 1) {
    if (tid < s) red[tid] = fmaxf(red[tid], red[tid + s]);
    __syncthreads();
  }
  mx = red[0];
  __syncthreads();

  float sum = 0.0f;
  #pragma unroll
  for (int i = 0; i < LL / 256; ++i) {
    v[i] = expf(v[i] - mx);
    sum += v[i];
  }
  red[tid] = sum;
  __syncthreads();
  for (int s = 128; s > 0; s >>= 1) {
    if (tid < s) red[tid] += red[tid + s];
    __syncthreads();
  }
  const float inv = 1.0f / red[0];

  #pragma unroll
  for (int i = 0; i < LL / 256; ++i)
    row[tid + i * 256] = v[i] * inv + mrow[tid + i * 256];
}

extern "C" void kernel_launch(void* const* d_in, const int* in_sizes, int n_in,
                              void* d_out, int out_size, void* d_ws, size_t ws_size,
                              hipStream_t stream) {
  const float* inputs = (const float*)d_in[0];  // [B,L,H]
  const float* mask   = (const float*)d_in[1];  // [B,L,L]
  const float* W      = (const float*)d_in[2];  // [H,H]
  const float* bvec   = (const float*)d_in[3];  // [H]

  float* hout = (float*)d_out;                        // [B,L,H]
  float* aout = hout + (size_t)BB * LL * HH;          // [B,L,L]
  float* q    = (float*)d_ws;                         // [B,L,H] fp32 scratch

  // 1) q = inputs @ W^T + b     (M=B*L, N=H, K=H; Bt = W row-major [N][K])
  gemm_splitf16_kernel<0><<<dim3(HH / 64, (BB * LL) / 128, 1), 256, 0, stream>>>(
      inputs, W, bvec, q, BB * LL, HH, HH, HH, HH, HH, 0L, 0L, 0L);

  // 2) u = q @ inputs^T per batch (M=L, N=L, K=H; Bt = inputs row-major [N][K])
  gemm_splitf16_kernel<0><<<dim3(LL / 64, LL / 128, BB), 256, 0, stream>>>(
      q, inputs, nullptr, aout, LL, LL, HH, HH, HH, LL,
      (long)LL * HH, (long)LL * HH, (long)LL * LL);

  // 3) a = softmax(relu(u)) + mask, in place in the output 'a' region
  softmax_relu_mask_kernel<<<dim3(BB * LL), 256, 0, stream>>>(aout, mask);

  // 4) h = a @ inputs per batch (M=L, N=H, K=L; B = inputs row-major [K][N])
  gemm_splitf16_kernel<1><<<dim3(HH / 64, LL / 128, BB), 256, 0, stream>>>(
      aout, inputs, nullptr, hout, LL, HH, LL, LL, HH, HH,
      (long)LL * LL, (long)LL * HH, (long)LL * HH);
}
